// SparseMultiHeadAttention_29119878267148
// MI455X (gfx1250) — compile-verified, matrix operand order checked by reading, not run on hardware
//
#include <hip/hip_runtime.h>
#include <hip/hip_bf16.h>

// ---------------------------------------------------------------------------
// Sparse multi-head attention (MoE head routing) for MI455X / gfx1250.
// bf16 WMMA (v_wmma_f32_16x16x32_bf16) for all GEMM-shaped work.
// LDS-staged operand pipelines via gfx1250 async global->LDS (ASYNCcnt),
// unroll-by-2 software pipelining with static double-buffer indices, and
// one-tile-ahead ds_load rotation inside the WMMA tile loops.
// ---------------------------------------------------------------------------

typedef __bf16 bf16_t;
typedef __bf16 v16bf __attribute__((ext_vector_type(16)));
typedef __bf16 v8bf  __attribute__((ext_vector_type(8)));
typedef float  v8f   __attribute__((ext_vector_type(8)));
typedef int    v4i_t __attribute__((ext_vector_type(4)));

constexpr int B_  = 8;
constexpr int S_  = 1024;
constexpr int D_  = 1024;
constexpr int H_  = 16;
constexpr int A_  = 4;
constexpr int DK_ = 128;
constexpr int F_  = A_ * DK_;   // 512

#if defined(__HIP_DEVICE_COMPILE__) && __has_builtin(__builtin_amdgcn_global_load_async_to_lds_b128)
#define USE_ASYNC 1
#if __has_builtin(__builtin_amdgcn_s_wait_asynccnt)
#define WAIT_ASYNC() __builtin_amdgcn_s_wait_asynccnt(0)
#else
#define WAIT_ASYNC() asm volatile("s_wait_asynccnt 0x0" ::: "memory")
#endif
// Builtin prototype: (v4i AS1* src, v4i AS3* dst, imm offset, imm cpol)
#define AS1(p) ((__attribute__((address_space(1))) v4i_t*)(p))
#define AS3(p) ((__attribute__((address_space(3))) v4i_t*)(p))
#else
#define USE_ASYNC 0
#define WAIT_ASYNC()
#endif

__device__ __forceinline__ v8f vzero8() {
  v8f v;
#pragma unroll
  for (int i = 0; i < 8; ++i) v[i] = 0.0f;
  return v;
}

// Per-lane 16-element bf16 operand fragment. Elements 0..7 from p, 8..15 from
// p+16 (ISA 16-bit A/B layout: lanes 0-15 hold K 0..7 & 16..23, lanes 16-31
// hold K 8..15 & 24..31; caller bakes the (lane>>4)*8 chunk offset into p).
__device__ __forceinline__ v16bf load_ab(const bf16_t* p) {
  union { v16bf v; v8bf h[2]; } u;
  u.h[0] = *reinterpret_cast<const v8bf*>(p);
  u.h[1] = *reinterpret_cast<const v8bf*>(p + 16);
  return u.v;
}

__device__ __forceinline__ v8f wmma_bf16(v16bf a, v16bf b, v8f c) {
  return __builtin_amdgcn_wmma_f32_16x16x32_bf16(
      false, a, false, b, (short)0, c, false, false);
}

// ---------------------------------------------------------------------------
// 1) fp32 -> bf16 cast / transpose
// ---------------------------------------------------------------------------
__global__ void cast_f32_bf16(const float* __restrict__ src,
                              bf16_t* __restrict__ dst, int n) {
  int i = blockIdx.x * blockDim.x + threadIdx.x;
  if (i < n) dst[i] = (bf16_t)src[i];
}

__global__ void trans_f32_bf16(const float* __restrict__ src,
                               bf16_t* __restrict__ dst, int rows, int cols) {
  size_t per = (size_t)rows * cols;
  src += (size_t)blockIdx.y * per;
  dst += (size_t)blockIdx.y * per;
  int gid = blockIdx.x * blockDim.x + threadIdx.x;   // gid = c*rows + r
  if (gid < (int)per) {
    int c = gid / rows;
    int r = gid - c * rows;
    dst[gid] = (bf16_t)src[(size_t)r * cols + c];
  }
}

// ---------------------------------------------------------------------------
// 2) Router
// ---------------------------------------------------------------------------
__global__ void xbar_kernel(const float* __restrict__ x,
                            float* __restrict__ xbar) {
  int gid = blockIdx.x * blockDim.x + threadIdx.x;   // 0..B*D-1
  int b = gid >> 10;
  int d = gid & (D_ - 1);
  float acc = 0.0f;
  for (int s = 0; s < S_; ++s)
    acc += x[((size_t)b * S_ + s) * D_ + d];
  xbar[gid] = acc * (1.0f / (float)S_);
}

__global__ void router_kernel(const float* __restrict__ xbar,
                              const float* __restrict__ Wr,
                              const float* __restrict__ br,
                              int* __restrict__ idx,
                              float* __restrict__ gates) {
  __shared__ float lg[B_][H_];
  int t = threadIdx.x;
  if (t < B_ * H_) {
    int b = t >> 4, h = t & 15;
    float acc = br[h];
    for (int d = 0; d < D_; ++d)
      acc += xbar[b * D_ + d] * Wr[d * H_ + h];
    lg[b][h] = acc;
  }
  __syncthreads();
  if (t < B_) {
    float p[H_];
    float mx = lg[t][0];
    for (int h = 1; h < H_; ++h) mx = fmaxf(mx, lg[t][h]);
    float den = 0.0f;
    for (int h = 0; h < H_; ++h) { p[h] = __expf(lg[t][h] - mx); den += p[h]; }
    for (int h = 0; h < H_; ++h) p[h] /= den;
    int   id[A_];
    float val[A_];
    for (int a = 0; a < A_; ++a) {
      int best = 0; float bv = -1e30f;
      for (int h = 0; h < H_; ++h)
        if (p[h] > bv) { bv = p[h]; best = h; }
      id[a] = best; val[a] = bv; p[best] = -1e30f;
    }
    float sp[H_];
    for (int h = 0; h < H_; ++h) sp[h] = 0.0f;
    for (int a = 0; a < A_; ++a) sp[id[a]] = val[a];
    float m2 = sp[0];
    for (int h = 1; h < H_; ++h) m2 = fmaxf(m2, sp[h]);
    float d2 = 0.0f;
    for (int h = 0; h < H_; ++h) d2 += __expf(sp[h] - m2);
    for (int a = 0; a < A_; ++a) {
      idx[t * A_ + a]   = id[a];
      gates[t * A_ + a] = __expf(val[a] - m2) / d2;
    }
  }
}

// ---------------------------------------------------------------------------
// 3) QKV projection: [S,D] x Wt[h] ([Dk-major, D]) -> Q/K row-major, V^T
//    grid: x = row-block (S/128), y = (b*A + a)*3 + which; 256 thr / 8 waves.
//    Weight slab (128n x 32k) staged in LDS, double-buffered, unrolled x2.
// ---------------------------------------------------------------------------
__global__ __launch_bounds__(256)
void qkv_gemm(const bf16_t* __restrict__ xbf,
              const bf16_t* __restrict__ wqt, const bf16_t* __restrict__ wkt,
              const bf16_t* __restrict__ wvt,
              const float* __restrict__ bq, const float* __restrict__ bk,
              const float* __restrict__ bv,
              const int* __restrict__ idx,
              bf16_t* __restrict__ qb, bf16_t* __restrict__ kb,
              bf16_t* __restrict__ vtb) {
  __shared__ bf16_t wlds[2][128][40];   // 40 = 32 + pad (bank rotate)

  int mat = blockIdx.y;
  int bb = mat / (A_ * 3);
  int rem = mat % (A_ * 3);
  int aa = rem / 3;
  int which = rem % 3;
  int h = idx[bb * A_ + aa];

  const bf16_t* wt = (which == 0 ? wqt : which == 1 ? wkt : wvt) +
                     (size_t)h * DK_ * D_;
  const float* bias = (which == 0 ? bq : which == 1 ? bk : bv) + h * DK_;

  int tid = threadIdx.x;
  int wid = tid >> 5;
  int lane = tid & 31;
  int l16 = lane & 15;
  int ch = (lane >> 4) * 8;
  int row0 = blockIdx.x * 128 + wid * 16;

  const bf16_t* arow =
      xbf + (size_t)bb * S_ * D_ + (size_t)(row0 + l16) * D_ + ch;

  v8f c[8];
#pragma unroll
  for (int t = 0; t < 8; ++t) c[t] = vzero8();

  // One-tile-ahead ds_load rotation so WMMA t overlaps the loads for t+1.
  auto compute = [&](int buf, v16bf av) {
    v16bf bt = load_ab(&wlds[buf][l16][ch]);
#pragma unroll
    for (int t = 0; t < 8; ++t) {
      v16bf btn = bt;
      if (t < 7) btn = load_ab(&wlds[buf][(t + 1) * 16 + l16][ch]);
      c[t] = wmma_bf16(av, bt, c[t]);
      bt = btn;
    }
  };

#if USE_ASYNC
  auto issue = [&](int buf, int k0) {
#pragma unroll
    for (int q = 0; q < 2; ++q) {
      int cc = q * 256 + tid;          // 512 chunks of 16B
      int n = cc >> 2;
      int off = (cc & 3) * 8;          // elems
      __builtin_amdgcn_global_load_async_to_lds_b128(
          AS1(wt + (size_t)n * D_ + k0 + off), AS3(&wlds[buf][n][off]), 0, 0);
    }
  };
  issue(0, 0);
  v16bf av = load_ab(arow);
#pragma unroll 1
  for (int i = 0; i < 30; i += 2) {
    WAIT_ASYNC();
    __syncthreads();
    issue(1, (i + 1) * 32);
    v16bf avn = load_ab(arow + (i + 1) * 32);
    compute(0, av);
    WAIT_ASYNC();
    __syncthreads();
    issue(0, (i + 2) * 32);
    av = load_ab(arow + (i + 2) * 32);
    compute(1, avn);
  }
  WAIT_ASYNC();
  __syncthreads();
  issue(1, 31 * 32);
  v16bf avl = load_ab(arow + 31 * 32);
  compute(0, av);
  WAIT_ASYNC();
  __syncthreads();
  compute(1, avl);
#else
  int ln = tid >> 1;
  int lk = (tid & 1) * 16;
  const bf16_t* wrow = wt + (size_t)ln * D_ + lk;
  v8bf w0 = *(const v8bf*)(wrow);
  v8bf w1 = *(const v8bf*)(wrow + 8);
  *(v8bf*)&wlds[0][ln][lk]     = w0;
  *(v8bf*)&wlds[0][ln][lk + 8] = w1;
  v16bf av = load_ab(arow);
  for (int i = 0; i < 32; ++i) {
    v8bf nw0 = w0, nw1 = w1;
    v16bf avn = av;
    if (i < 31) {
      nw0 = *(const v8bf*)(wrow + (i + 1) * 32);
      nw1 = *(const v8bf*)(wrow + (i + 1) * 32 + 8);
      avn = load_ab(arow + (i + 1) * 32);
    }
    __syncthreads();
    compute(i & 1, av);
    if (i < 31) {
      *(v8bf*)&wlds[(i + 1) & 1][ln][lk]     = nw0;
      *(v8bf*)&wlds[(i + 1) & 1][ln][lk + 8] = nw1;
      av = avn;
    }
  }
#endif

  int pair = bb * A_ + aa;
#pragma unroll
  for (int t = 0; t < 8; ++t) {
    int col = t * 16 + l16;
    float bsv = bias[col];
#pragma unroll
    for (int r = 0; r < 8; ++r) {
      int row = row0 + r + ((lane >> 4) << 3);
      float v = c[t][r] + bsv;
      if (which == 0)
        qb[((size_t)pair * S_ + row) * DK_ + col] = (bf16_t)v;
      else if (which == 1)
        kb[((size_t)pair * S_ + row) * DK_ + col] = (bf16_t)v;
      else
        vtb[((size_t)pair * DK_ + col) * S_ + row] = (bf16_t)v;
    }
  }
}

// ---------------------------------------------------------------------------
// 4) Flash-style attention per (b, active-head). K/V slabs (32 keys) staged
//    in LDS (shared by the block's 4 waves), double-buffered, unrolled x2;
//    P relaid C->A through per-wave LDS. grid: x = q-block, y = pair; 128 thr.
// ---------------------------------------------------------------------------
__global__ __launch_bounds__(128)
void attn_kernel(const bf16_t* __restrict__ qb, const bf16_t* __restrict__ kbuf,
                 const bf16_t* __restrict__ vtb,
                 const float* __restrict__ gates,
                 bf16_t* __restrict__ cat) {
  __shared__ bf16_t klds[2][32][136];   // [key][dk], padded
  __shared__ bf16_t vlds[2][128][40];   // [dk][key], padded
  __shared__ bf16_t plds[4][16][40];    // per-wave P staging, padded

  int pair = blockIdx.y;
  int bb = pair >> 2;
  int aa = pair & 3;
  int tid = threadIdx.x;
  int wid = tid >> 5;
  int lane = tid & 31;
  int l16 = lane & 15;
  int ch = (lane >> 4) * 8;
  int q0 = blockIdx.x * 64 + wid * 16;

  const bf16_t* Q  = qb   + (size_t)pair * S_ * DK_;
  const bf16_t* K  = kbuf + (size_t)pair * S_ * DK_;
  const bf16_t* Vt = vtb  + (size_t)pair * DK_ * S_;

  v16bf qa[4];
#pragma unroll
  for (int kk = 0; kk < 4; ++kk)
    qa[kk] = load_ab(Q + (size_t)(q0 + l16) * DK_ + kk * 32 + ch);

  float m[8], l[8];
  v8f o[8];
#pragma unroll
  for (int r = 0; r < 8; ++r) { m[r] = -1e30f; l[r] = 0.0f; }
#pragma unroll
  for (int t = 0; t < 8; ++t) o[t] = vzero8();

  const float scale = 0.08838834764831845f;  // 1/sqrt(128)

  auto compute_step = [&](int cur) {
    v8f s0 = vzero8(), s1 = vzero8();
    v16bf k0 = load_ab(&klds[cur][l16][ch]);
    v16bf k1 = load_ab(&klds[cur][16 + l16][ch]);
#pragma unroll
    for (int kk = 0; kk < 4; ++kk) {
      v16bf k0n = k0, k1n = k1;
      if (kk < 3) {
        k0n = load_ab(&klds[cur][l16][(kk + 1) * 32 + ch]);
        k1n = load_ab(&klds[cur][16 + l16][(kk + 1) * 32 + ch]);
      }
      s0 = wmma_bf16(qa[kk], k0, s0);
      s1 = wmma_bf16(qa[kk], k1, s1);
      k0 = k0n;
      k1 = k1n;
    }
    float p0[8], p1[8], corr[8];
#pragma unroll
    for (int r = 0; r < 8; ++r) {
      float a0 = s0[r] * scale;
      float a1 = s1[r] * scale;
      float rm = fmaxf(a0, a1);
#pragma unroll
      for (int off = 1; off < 16; off <<= 1)
        rm = fmaxf(rm, __shfl_xor(rm, off, 32));
      float mn = fmaxf(m[r], rm);
      corr[r] = __expf(m[r] - mn);
      m[r] = mn;
      float e0 = __expf(a0 - mn);
      float e1 = __expf(a1 - mn);
      float rs = e0 + e1;
#pragma unroll
      for (int off = 1; off < 16; off <<= 1)
        rs += __shfl_xor(rs, off, 32);
      l[r] = l[r] * corr[r] + rs;
      p0[r] = e0;
      p1[r] = e1;
    }
#pragma unroll
    for (int t = 0; t < 8; ++t)
#pragma unroll
      for (int r = 0; r < 8; ++r) o[t][r] *= corr[r];
    // C-layout -> per-wave LDS row-major bf16 (same-wave DS stays in order)
#pragma unroll
    for (int r = 0; r < 8; ++r) {
      int row = r + ((lane >> 4) << 3);
      plds[wid][row][l16]      = (bf16_t)p0[r];
      plds[wid][row][l16 + 16] = (bf16_t)p1[r];
    }
    v16bf pa = load_ab(&plds[wid][l16][ch]);
    v16bf vb = load_ab(&vlds[cur][l16][ch]);
#pragma unroll
    for (int t = 0; t < 8; ++t) {
      v16bf vbn = vb;
      if (t < 7) vbn = load_ab(&vlds[cur][(t + 1) * 16 + l16][ch]);
      o[t] = wmma_bf16(pa, vb, o[t]);
      vb = vbn;
    }
  };

#if USE_ASYNC
  auto issue = [&](int buf, int jb) {
#pragma unroll
    for (int q = 0; q < 4; ++q) {
      int cc = q * 128 + tid;               // 512 chunks of 16B each
      int kr = cc >> 4;                     // K: key row, 16 chunks/row
      int ko = (cc & 15) * 8;               // elems
      __builtin_amdgcn_global_load_async_to_lds_b128(
          AS1(K + (size_t)(jb + kr) * DK_ + ko), AS3(&klds[buf][kr][ko]), 0, 0);
      int vn = cc >> 2;                     // V: dk row, 4 chunks/row
      int vo = (cc & 3) * 8;
      __builtin_amdgcn_global_load_async_to_lds_b128(
          AS1(Vt + (size_t)vn * S_ + jb + vo), AS3(&vlds[buf][vn][vo]), 0, 0);
    }
  };
  issue(0, 0);
#pragma unroll 1
  for (int it = 0; it < 30; it += 2) {
    WAIT_ASYNC();
    __syncthreads();
    issue(1, (it + 1) * 32);
    compute_step(0);
    WAIT_ASYNC();
    __syncthreads();
    issue(0, (it + 2) * 32);
    compute_step(1);
  }
  WAIT_ASYNC();
  __syncthreads();
  issue(1, 31 * 32);
  compute_step(0);
  WAIT_ASYNC();
  __syncthreads();
  compute_step(1);
#else
  {
    int kr = tid >> 2, ko = (tid & 3) * 32;
#pragma unroll
    for (int j = 0; j < 4; ++j) {
      *(v8bf*)&klds[0][kr][ko + j * 8] =
          *(const v8bf*)(K + (size_t)kr * DK_ + ko + j * 8);
      *(v8bf*)&vlds[0][tid][j * 8] =
          *(const v8bf*)(Vt + (size_t)tid * S_ + j * 8);
    }
  }
  for (int it = 0; it < 32; ++it) {
    int jb = it * 32;
    v8bf kpre[4], vpre[4];
    int kr = tid >> 2, ko = (tid & 3) * 32;
    if (it < 31) {
#pragma unroll
      for (int j = 0; j < 4; ++j) {
        kpre[j] = *(const v8bf*)(K + (size_t)(jb + 32 + kr) * DK_ + ko + j * 8);
        vpre[j] = *(const v8bf*)(Vt + (size_t)tid * S_ + jb + 32 + j * 8);
      }
    }
    __syncthreads();
    compute_step(it & 1);
    if (it < 31) {
#pragma unroll
      for (int j = 0; j < 4; ++j) {
        *(v8bf*)&klds[(it + 1) & 1][kr][ko + j * 8] = kpre[j];
        *(v8bf*)&vlds[(it + 1) & 1][tid][j * 8]     = vpre[j];
      }
    }
  }
#endif

  float g = gates[pair];
  float inv[8];
#pragma unroll
  for (int r = 0; r < 8; ++r) inv[r] = g / l[r];

#pragma unroll
  for (int t = 0; t < 8; ++t) {
    int col = aa * DK_ + t * 16 + l16;
#pragma unroll
    for (int r = 0; r < 8; ++r) {
      int row = q0 + r + ((lane >> 4) << 3);
      cat[((size_t)bb * S_ + row) * F_ + col] = (bf16_t)(o[t][r] * inv[r]);
    }
  }
}

// ---------------------------------------------------------------------------
// 5) Output projection: [B*S, 512] x WoT([1024, 512]) + bo -> fp32 out
//    grid: x = row-block (64), y = col-block (8); WoT slab staged in LDS.
// ---------------------------------------------------------------------------
__global__ __launch_bounds__(256)
void out_gemm(const bf16_t* __restrict__ cat, const bf16_t* __restrict__ wot,
              const float* __restrict__ bo, float* __restrict__ out) {
  __shared__ bf16_t wlds[2][128][40];

  int tid = threadIdx.x;
  int wid = tid >> 5;
  int lane = tid & 31;
  int l16 = lane & 15;
  int ch = (lane >> 4) * 8;
  int row0 = blockIdx.x * 128 + wid * 16;
  int col0 = blockIdx.y * 128;

  const bf16_t* wbase = wot + (size_t)col0 * F_;
  const bf16_t* arow = cat + (size_t)(row0 + l16) * F_ + ch;

  v8f c[8];
#pragma unroll
  for (int t = 0; t < 8; ++t) c[t] = vzero8();

  auto compute = [&](int buf, v16bf av) {
    v16bf bt = load_ab(&wlds[buf][l16][ch]);
#pragma unroll
    for (int t = 0; t < 8; ++t) {
      v16bf btn = bt;
      if (t < 7) btn = load_ab(&wlds[buf][(t + 1) * 16 + l16][ch]);
      c[t] = wmma_bf16(av, bt, c[t]);
      bt = btn;
    }
  };

#if USE_ASYNC
  auto issue = [&](int buf, int k0) {
#pragma unroll
    for (int q = 0; q < 2; ++q) {
      int cc = q * 256 + tid;
      int n = cc >> 2;
      int off = (cc & 3) * 8;
      __builtin_amdgcn_global_load_async_to_lds_b128(
          AS1(wbase + (size_t)n * F_ + k0 + off), AS3(&wlds[buf][n][off]), 0, 0);
    }
  };
  issue(0, 0);
  v16bf av = load_ab(arow);
#pragma unroll 1
  for (int i = 0; i < 14; i += 2) {
    WAIT_ASYNC();
    __syncthreads();
    issue(1, (i + 1) * 32);
    v16bf avn = load_ab(arow + (i + 1) * 32);
    compute(0, av);
    WAIT_ASYNC();
    __syncthreads();
    issue(0, (i + 2) * 32);
    av = load_ab(arow + (i + 2) * 32);
    compute(1, avn);
  }
  WAIT_ASYNC();
  __syncthreads();
  issue(1, 15 * 32);
  v16bf avl = load_ab(arow + 15 * 32);
  compute(0, av);
  WAIT_ASYNC();
  __syncthreads();
  compute(1, avl);
#else
  int ln = tid >> 1;
  int lk = (tid & 1) * 16;
  const bf16_t* wrow = wbase + (size_t)ln * F_ + lk;
  v8bf w0 = *(const v8bf*)(wrow);
  v8bf w1 = *(const v8bf*)(wrow + 8);
  *(v8bf*)&wlds[0][ln][lk]     = w0;
  *(v8bf*)&wlds[0][ln][lk + 8] = w1;
  v16bf av = load_ab(arow);
  for (int i = 0; i < 16; ++i) {
    v8bf nw0 = w0, nw1 = w1;
    v16bf avn = av;
    if (i < 15) {
      nw0 = *(const v8bf*)(wrow + (i + 1) * 32);
      nw1 = *(const v8bf*)(wrow + (i + 1) * 32 + 8);
      avn = load_ab(arow + (i + 1) * 32);
    }
    __syncthreads();
    compute(i & 1, av);
    if (i < 15) {
      *(v8bf*)&wlds[(i + 1) & 1][ln][lk]     = nw0;
      *(v8bf*)&wlds[(i + 1) & 1][ln][lk + 8] = nw1;
      av = avn;
    }
  }
#endif

#pragma unroll
  for (int t = 0; t < 8; ++t) {
    int col = col0 + t * 16 + l16;
    float bb = bo[col];
#pragma unroll
    for (int r = 0; r < 8; ++r) {
      int row = row0 + r + ((lane >> 4) << 3);
      out[(size_t)row * D_ + col] = c[t][r] + bb;
    }
  }
}

// ---------------------------------------------------------------------------
// Launch
// ---------------------------------------------------------------------------
extern "C" void kernel_launch(void* const* d_in, const int* in_sizes, int n_in,
                              void* d_out, int out_size, void* d_ws,
                              size_t ws_size, hipStream_t stream) {
  const float* x  = (const float*)d_in[0];
  const float* Wq = (const float*)d_in[1];
  const float* bq = (const float*)d_in[2];
  const float* Wk = (const float*)d_in[3];
  const float* bk = (const float*)d_in[4];
  const float* Wv = (const float*)d_in[5];
  const float* bv = (const float*)d_in[6];
  const float* Wr = (const float*)d_in[7];
  const float* br = (const float*)d_in[8];
  const float* Wo = (const float*)d_in[9];
  const float* bo = (const float*)d_in[10];
  float* out = (float*)d_out;

  char* p = (char*)d_ws;
  size_t nx   = (size_t)B_ * S_ * D_;
  size_t nwh  = (size_t)H_ * DK_ * D_;
  size_t nwo  = (size_t)D_ * F_;
  size_t nqkv = (size_t)B_ * A_ * S_ * DK_;
  size_t ncat = (size_t)B_ * S_ * F_;

  bf16_t* xbf = (bf16_t*)p;            p += nx * 2;
  bf16_t* wqt = (bf16_t*)p;            p += nwh * 2;
  bf16_t* wkt = (bf16_t*)p;            p += nwh * 2;
  bf16_t* wvt = (bf16_t*)p;            p += nwh * 2;
  bf16_t* wot = (bf16_t*)p;            p += nwo * 2;
  bf16_t* qb  = (bf16_t*)p;            p += nqkv * 2;
  bf16_t* kb  = (bf16_t*)p;            p += nqkv * 2;
  bf16_t* vtb = (bf16_t*)p;            p += nqkv * 2;
  bf16_t* cat = (bf16_t*)p;            p += ncat * 2;
  float*  xbar  = (float*)p;           p += (size_t)B_ * D_ * 4;
  float*  gates = (float*)p;           p += 256;
  int*    idx   = (int*)p;             p += 256;

  cast_f32_bf16<<<(int)(nx / 256), 256, 0, stream>>>(x, xbf, (int)nx);
  trans_f32_bf16<<<dim3((D_ * DK_ + 255) / 256, H_), 256, 0, stream>>>(
      Wq, wqt, D_, DK_);
  trans_f32_bf16<<<dim3((D_ * DK_ + 255) / 256, H_), 256, 0, stream>>>(
      Wk, wkt, D_, DK_);
  trans_f32_bf16<<<dim3((D_ * DK_ + 255) / 256, H_), 256, 0, stream>>>(
      Wv, wvt, D_, DK_);
  trans_f32_bf16<<<dim3((F_ * D_ + 255) / 256, 1), 256, 0, stream>>>(
      Wo, wot, F_, D_);

  xbar_kernel<<<(B_ * D_) / 256, 256, 0, stream>>>(x, xbar);
  router_kernel<<<1, 256, 0, stream>>>(xbar, Wr, br, idx, gates);

  qkv_gemm<<<dim3(S_ / 128, B_ * A_ * 3), 256, 0, stream>>>(
      xbf, wqt, wkt, wvt, bq, bk, bv, idx, qb, kb, vtb);

  attn_kernel<<<dim3(S_ / 64, B_ * A_), 128, 0, stream>>>(
      qb, kb, vtb, gates, cat);

  out_gemm<<<dim3((B_ * S_) / 128, D_ / 128), 256, 0, stream>>>(
      cat, wot, bo, out);

  (void)in_sizes; (void)n_in; (void)out_size; (void)ws_size;
}
